// NeRF_5815385719207
// MI455X (gfx1250) — matrix operand order repaired and checked by reading, compiled, not run on hardware
//
#include <hip/hip_runtime.h>
#include <hip/hip_bf16.h>

typedef __attribute__((ext_vector_type(16))) _Float16 v16h;
typedef __attribute__((ext_vector_type(8)))  _Float16 v8h;
typedef __attribute__((ext_vector_type(8)))  float    v8f;

#define N_PTS_TOTAL 524288
#define SAMP 4
#define PTS 32
#define ROWS 128           // PTS*SAMP
#define THREADS 128
#define NWAVES 4

// ---- workspace layout (units: _Float16) ----
#define WS_WG   0                   // glob_w^T-rows  [32][64]  (57 padded -> 64)
#define WS_WXV  (WS_WG  + 32*64)    // col1_w[:,0:88] [64][96]  (88 padded -> 96)
#define WS_WR   (WS_WXV + 64*96)    // col1_w[:,88:]  [64][32]  (23 padded -> 32)
#define WS_WL0  (WS_WR  + 64*32)    // lr0_w          [64][32]  (24 padded -> 32)
#define WS_WFC  (WS_WL0 + 64*32)    // fc_w           [16][32]

// ---- LDS layout (bytes, all 16B aligned) ----
#define SM_FEATA 0        // [128][64] f16 = 16384  (reused as H16 in phase 6/7)
#define SM_GF    16384    // [128][32] f16 = 8192
#define SM_RAWA  24576    // [128][32] f16 = 8192   (raw 23ch padded to 32, A layout)
#define SM_RGB   32768    // [128][3]  f32 = 1536
#define SM_VIF   34304    // [32][32]  f16 = 2048   (vox8 | img_feat16 | pad8)
#define SM_IMA   36352    // [32][32]  f16 = 2048   (im_feat A-tile)
#define SM_X32   38400    // [32][64]  f32 = 8192   (trunk x, fp32 for sigma)
#define SM_XVA   46592    // [32][96]  f16 = 6144   ([x | vif] A-tile)
#define SM_HP    52736    // [32][64]  f32 = 8192   (per-point partial of col1)
#define SM_TOTAL 60928

__device__ __forceinline__ v8f wmma_f16(v16h a, v16h b, v8f c) {
  return __builtin_amdgcn_wmma_f32_16x16x32_f16(false, a, false, b, (short)0, c,
                                                false, false);
}

// A fragment (16-bit, 16x32 MxK): lane holds M=lane%16; half-wave selects K 0..15 / 16..31.
__device__ __forceinline__ v16h load_a_frag(const _Float16* A, int strideH,
                                            int rowTile, int ks, int lane) {
  int m  = rowTile * 16 + (lane & 15);
  int hi = (lane >> 4) & 1;
  const _Float16* p = A + m * strideH + ks * 32 + hi * 16;
  return *(const v16h*)p;
}

// B fragment (16-bit, 32x16 KxN): lane = column N; half-wave selects K 0..15 / 16..31.
// W stored row-major [Nout][Kpad] so this is one contiguous 32B load per lane.
__device__ __forceinline__ v16h load_b_frag(const _Float16* W, int Kpad,
                                            int nTile, int ks, int lane) {
  int n  = nTile * 16 + (lane & 15);
  int kh = (lane >> 4) & 1;
  return *(const v16h*)(W + n * Kpad + ks * 32 + kh * 16);
}

__global__ void prep_weights(const float* __restrict__ gw,
                             const float* __restrict__ c1w,
                             const float* __restrict__ l0w,
                             const float* __restrict__ fcw,
                             _Float16* __restrict__ ws) {
  int t = threadIdx.x;
  for (int i = t; i < 32 * 64; i += 256) {               // glob [32][57] -> [32][64]
    int n = i >> 6, k = i & 63;
    ws[WS_WG + i] = (_Float16)(k < 57 ? gw[n * 57 + k] : 0.f);
  }
  for (int i = t; i < 64 * 96; i += 256) {               // col1[:,0:88] -> [64][96]
    int n = i / 96, k = i % 96;
    ws[WS_WXV + i] = (_Float16)(k < 88 ? c1w[n * 111 + k] : 0.f);
  }
  for (int i = t; i < 64 * 32; i += 256) {               // col1[:,88:111] -> [64][32]
    int n = i >> 5, k = i & 31;
    ws[WS_WR + i] = (_Float16)(k < 23 ? c1w[n * 111 + 88 + k] : 0.f);
  }
  for (int i = t; i < 64 * 32; i += 256) {               // lr0 [64][24] -> [64][32]
    int n = i >> 5, k = i & 31;
    ws[WS_WL0 + i] = (_Float16)(k < 24 ? l0w[n * 24 + k] : 0.f);
  }
  for (int i = t; i < 16 * 32; i += 256)                 // fc [16][32]
    ws[WS_WFC + i] = (_Float16)fcw[i];
}

__global__ __launch_bounds__(THREADS)
void nerf_fused(const float* __restrict__ vox,      // [N,8]
                const float* __restrict__ img,      // [N,S,23]
                const float* __restrict__ view_w, const float* __restrict__ view_b,
                const float* __restrict__ glob_b,
                const float* __restrict__ aggw_w, const float* __restrict__ aggw_b,
                const float* __restrict__ fc_b,
                const float* __restrict__ lr0_b,
                const float* __restrict__ sigma_w, const float* __restrict__ sigma_b,
                const float* __restrict__ col1_b,
                const float* __restrict__ col2_w, const float* __restrict__ col2_b,
                const _Float16* __restrict__ ws,
                float* __restrict__ out) {          // [N,4]
  __shared__ __align__(64) char smem[SM_TOTAL];
  _Float16* featA = (_Float16*)(smem + SM_FEATA);
  _Float16* h16   = (_Float16*)(smem + SM_FEATA);   // region reuse (featA dead)
  _Float16* gf16  = (_Float16*)(smem + SM_GF);
  _Float16* rawA  = (_Float16*)(smem + SM_RAWA);
  float*    rgbF  = (float*)   (smem + SM_RGB);
  _Float16* vifh  = (_Float16*)(smem + SM_VIF);
  _Float16* imA   = (_Float16*)(smem + SM_IMA);
  float*    x32   = (float*)   (smem + SM_X32);
  _Float16* xvA   = (_Float16*)(smem + SM_XVA);
  float*    hp32  = (float*)   (smem + SM_HP);

  const int t    = threadIdx.x;
  const int lane = t & 31;
  const int w    = t >> 5;
  const int p0   = blockIdx.x * PTS;

  // ---------- Phase 1: view MLP + residual + var/mean over S -> featA, rawA ----------
  {
    const int r = t;                                   // one (pt,s) row per thread
    const long g = (long)p0 * SAMP + r;
    const float* rp = img + g * 23;
    float raw[23];
    #pragma unroll
    for (int c = 0; c < 23; ++c) raw[c] = rp[c];
    const float d0 = raw[19], d1 = raw[20], d2 = raw[21], d3 = raw[22];
    _Float16* fa = featA + r * 64;
    #pragma unroll
    for (int c = 0; c < 19; ++c) {
      float v = view_w[c*4+0]*d0 + view_w[c*4+1]*d1 +
                view_w[c*4+2]*d2 + view_w[c*4+3]*d3 + view_b[c];
      float x = raw[c] + fmaxf(v, 0.f);                // img_feat_rgb
      float s1 = x + __shfl_xor(x, 1, 32);             // sum over 4-lane group (S dim)
      float sm = s1 + __shfl_xor(s1, 2, 32);
      float mean = sm * 0.25f;
      float e = x - mean, sq = e * e;
      float q1 = sq + __shfl_xor(sq, 1, 32);
      float qs = q1 + __shfl_xor(q1, 2, 32);
      float var = qs * (1.f / 3.f);                    // ddof=1
      fa[c]      = (_Float16)x;
      fa[19 + c] = (_Float16)var;
      fa[38 + c] = (_Float16)mean;
    }
    #pragma unroll
    for (int c = 57; c < 64; ++c) fa[c] = (_Float16)0.f;
    _Float16* ra = rawA + r * 32;
    #pragma unroll
    for (int c = 0; c < 23; ++c) ra[c] = (_Float16)raw[c];
    #pragma unroll
    for (int c = 23; c < 32; ++c) ra[c] = (_Float16)0.f;
    rgbF[r*3+0] = raw[16]; rgbF[r*3+1] = raw[17]; rgbF[r*3+2] = raw[18];
    if ((r & 3) == 0) {                                // per-point vox -> vif[0:8]
      int pt = r >> 2;
      const float* vp = vox + (long)(p0 + pt) * 8;
      _Float16* vf = vifh + pt * 32;
      #pragma unroll
      for (int c = 0; c < 8; ++c) vf[c] = (_Float16)vp[c];
      #pragma unroll
      for (int c = 24; c < 32; ++c) vf[c] = (_Float16)0.f;
    }
  }
  __syncthreads();

  // ---------- Phase 2: glob GEMM [128x64]x[64x32] -> gf16 (WMMA) ----------
  for (int j = w; j < 16; j += NWAVES) {               // 8 rowTiles x 2 nTiles
    int rt = j >> 1, nt = j & 1;
    v8f acc = {};
    #pragma unroll
    for (int ks = 0; ks < 2; ++ks) {
      v16h a = load_a_frag(featA, 64, rt, ks, lane);
      v16h b = load_b_frag(ws + WS_WG, 64, nt, ks, lane);
      acc = wmma_f16(a, b, acc);
    }
    int n = nt * 16 + (lane & 15);
    float bias = glob_b[n];
    int mbase = rt * 16 + ((lane >> 4) << 3);
    #pragma unroll
    for (int v = 0; v < 8; ++v)
      gf16[(mbase + v) * 32 + n] = (_Float16)fmaxf(acc[v] + bias, 0.f);
  }
  __syncthreads();

  // ---------- Phase 3: agg softmax over S + im_feat -> imA ----------
  {
    const int r = t;
    const _Float16* gfr = gf16 + r * 32;
    float aw = aggw_b[0];
    #pragma unroll
    for (int i = 0; i < 32; ++i) aw += aggw_w[i] * (float)gfr[i];
    aw = fmaxf(aw, 0.f);
    float m1 = fmaxf(aw, __shfl_xor(aw, 1, 32));
    float mx = fmaxf(m1, __shfl_xor(m1, 2, 32));
    float e  = __expf(aw - mx);
    float s1 = e + __shfl_xor(e, 1, 32);
    float se = s1 + __shfl_xor(s1, 2, 32);
    float wn = e / se;
    int pt = r >> 2;
    #pragma unroll
    for (int i = 0; i < 32; ++i) {
      float v  = (float)gfr[i] * wn;
      float v1 = v + __shfl_xor(v, 1, 32);
      float vs = v1 + __shfl_xor(v1, 2, 32);
      if ((r & 3) == 0) imA[pt * 32 + i] = (_Float16)vs;
    }
  }
  __syncthreads();

  // ---------- Phase 4: fc GEMM [32x32]x[32x16] -> vif[8:24] (WMMA) ----------
  if (w < 2) {                                         // wave-uniform branch
    int rt = w;
    v8f acc = {};
    v16h a = load_a_frag(imA, 32, rt, 0, lane);
    v16h b = load_b_frag(ws + WS_WFC, 32, 0, 0, lane);
    acc = wmma_f16(a, b, acc);
    int n = lane & 15;
    float bias = fc_b[n];
    int mbase = rt * 16 + ((lane >> 4) << 3);
    #pragma unroll
    for (int v = 0; v < 8; ++v)
      vifh[(mbase + v) * 32 + 8 + n] = (_Float16)fmaxf(acc[v] + bias, 0.f);
  }
  __syncthreads();

  // ---------- Phase 5: lr0 GEMM [32x32]x[32x64] -> x32 (WMMA) ----------
  for (int j = w; j < 8; j += NWAVES) {                // 2 rowTiles x 4 nTiles
    int rt = j >> 2, nt = j & 3;
    v8f acc = {};
    v16h a = load_a_frag(vifh, 32, rt, 0, lane);
    v16h b = load_b_frag(ws + WS_WL0, 32, nt, 0, lane);
    acc = wmma_f16(a, b, acc);
    int n = nt * 16 + (lane & 15);
    float bias = lr0_b[n];
    int mbase = rt * 16 + ((lane >> 4) << 3);
    #pragma unroll
    for (int v = 0; v < 8; ++v)
      x32[(mbase + v) * 64 + n] = fmaxf(acc[v] + bias, 0.f);
  }
  __syncthreads();

  // ---------- Phase 5b: sigma (fp32) + assemble xvA = [x | vif | pad] ----------
  if (t < PTS) {
    const float* xp = x32 + t * 64;
    float sg = sigma_b[0];
    #pragma unroll
    for (int i = 0; i < 64; ++i) sg += sigma_w[i] * xp[i];
    float sp = (sg > 20.f) ? sg : log1pf(__expf(sg));  // softplus
    out[(long)(p0 + t) * 4 + 3] = sp;
  }
  for (int i = t; i < PTS * 96; i += THREADS) {
    int pt = i / 96, k = i % 96;
    _Float16 val;
    if (k < 64)      val = (_Float16)x32[pt * 64 + k];
    else if (k < 88) val = vifh[pt * 32 + (k - 64)];
    else             val = (_Float16)0.f;
    xvA[i] = val;
  }
  __syncthreads();

  // ---------- Phase 6a: col1 per-point part [32x96]x[96x64] -> hp32 (WMMA) ----------
  for (int j = w; j < 8; j += NWAVES) {                // 2 rowTiles x 4 nTiles, 3 ksteps
    int rt = j >> 2, nt = j & 3;
    v8f acc = {};
    #pragma unroll
    for (int ks = 0; ks < 3; ++ks) {
      v16h a = load_a_frag(xvA, 96, rt, ks, lane);
      v16h b = load_b_frag(ws + WS_WXV, 96, nt, ks, lane);
      acc = wmma_f16(a, b, acc);
    }
    int n = nt * 16 + (lane & 15);
    int mbase = rt * 16 + ((lane >> 4) << 3);
    #pragma unroll
    for (int v = 0; v < 8; ++v) hp32[(mbase + v) * 64 + n] = acc[v];
  }
  __syncthreads();

  // ---------- Phase 6b: col1 per-row part [128x32]x[32x64] + hp + bias -> h16 ----------
  for (int j = w; j < 32; j += NWAVES) {               // 8 rowTiles x 4 nTiles
    int rt = j >> 2, nt = j & 3;
    v8f acc = {};
    v16h a = load_a_frag(rawA, 32, rt, 0, lane);
    v16h b = load_b_frag(ws + WS_WR, 32, nt, 0, lane);
    acc = wmma_f16(a, b, acc);
    int n = nt * 16 + (lane & 15);
    float bias = col1_b[n];
    int mbase = rt * 16 + ((lane >> 4) << 3);
    #pragma unroll
    for (int v = 0; v < 8; ++v) {
      int r = mbase + v;
      float val = fmaxf(acc[v] + hp32[(r >> 2) * 64 + n] + bias, 0.f);
      h16[r * 64 + n] = (_Float16)val;                 // overlays dead featA region
    }
  }
  __syncthreads();

  // ---------- Phase 7: col2 + softmax over S + color ----------
  {
    const int r = t;
    const _Float16* hr = h16 + r * 64;
    float cw = col2_b[0];
    #pragma unroll
    for (int i = 0; i < 64; ++i) cw += col2_w[i] * (float)hr[i];
    cw = fmaxf(cw, 0.f);
    float m1 = fmaxf(cw, __shfl_xor(cw, 1, 32));
    float mx = fmaxf(m1, __shfl_xor(m1, 2, 32));
    float e  = __expf(cw - mx);
    float s1 = e + __shfl_xor(e, 1, 32);
    float se = s1 + __shfl_xor(s1, 2, 32);
    float wn = e / se;
    float c0 = rgbF[r*3+0] * wn, c1 = rgbF[r*3+1] * wn, c2 = rgbF[r*3+2] * wn;
    c0 += __shfl_xor(c0, 1, 32); c0 += __shfl_xor(c0, 2, 32);
    c1 += __shfl_xor(c1, 1, 32); c1 += __shfl_xor(c1, 2, 32);
    c2 += __shfl_xor(c2, 1, 32); c2 += __shfl_xor(c2, 2, 32);
    if ((r & 3) == 0) {
      long ob = (long)(p0 + (r >> 2)) * 4;
      out[ob + 0] = c0; out[ob + 1] = c1; out[ob + 2] = c2;
    }
  }
}

extern "C" void kernel_launch(void* const* d_in, const int* in_sizes, int n_in,
                              void* d_out, int out_size, void* d_ws, size_t ws_size,
                              hipStream_t stream) {
  const float* vox     = (const float*)d_in[0];
  const float* img     = (const float*)d_in[1];
  const float* view_w  = (const float*)d_in[2];
  const float* view_b  = (const float*)d_in[3];
  const float* glob_w  = (const float*)d_in[4];
  const float* glob_b  = (const float*)d_in[5];
  const float* aggw_w  = (const float*)d_in[6];
  const float* aggw_b  = (const float*)d_in[7];
  const float* fc_w    = (const float*)d_in[8];
  const float* fc_b    = (const float*)d_in[9];
  const float* lr0_w   = (const float*)d_in[10];
  const float* lr0_b   = (const float*)d_in[11];
  const float* sigma_w = (const float*)d_in[12];
  const float* sigma_b = (const float*)d_in[13];
  const float* col1_w  = (const float*)d_in[14];
  const float* col1_b  = (const float*)d_in[15];
  const float* col2_w  = (const float*)d_in[16];
  const float* col2_b  = (const float*)d_in[17];
  _Float16* ws = (_Float16*)d_ws;
  float* out = (float*)d_out;

  prep_weights<<<1, 256, 0, stream>>>(glob_w, col1_w, lr0_w, fc_w, ws);

  int nblocks = N_PTS_TOTAL / PTS;   // 16384
  nerf_fused<<<nblocks, THREADS, 0, stream>>>(
      vox, img, view_w, view_b, glob_b, aggw_w, aggw_b, fc_b, lr0_b,
      sigma_w, sigma_b, col1_b, col2_w, col2_b, ws, out);
}
